// AttentionCritic_12146167513201
// MI455X (gfx1250) — matrix operand order repaired and checked by reading, not compiled, not run on hardware
//
#include <hip/hip_runtime.h>
#include <hip/hip_bf16.h>

#define N_AG   8
#define BATCH  16384
#define S_DIM  128
#define A_DIM  32
#define H_DIM  256
#define KH_N   4
#define AD_N   64
#define M_ROWS (N_AG * BATCH)   // 131072

typedef __attribute__((ext_vector_type(16))) __bf16          v16bf;
typedef __attribute__((ext_vector_type(16))) unsigned short  v16u;
typedef __attribute__((ext_vector_type(8)))  float           v8f;
typedef __attribute__((ext_vector_type(8)))  unsigned short  u16x8;
typedef __attribute__((ext_vector_type(4)))  float           f32x4;

__device__ __forceinline__ unsigned short f2bf(float f) {
  unsigned u = __float_as_uint(f);
  u += 0x7FFFu + ((u >> 16) & 1u);      // round-to-nearest-even
  return (unsigned short)(u >> 16);
}
__device__ __forceinline__ float bf2f(unsigned short h) {
  return __uint_as_float(((unsigned)h) << 16);
}

// ---------------------------------------------------------------------------
// bf16 WMMA GEMM with double-buffered LDS and optional fused [K,1] head.
//   Out[M,Ncols](bf16) = act(A[M,Kd] * B[Kd,Ncols] + bias)            (store)
//   redOut[row] += redScale * sum_col( leaky(..) * Wred[col] )        (reduce)
//  - A: fp32 (A32) OR bf16 A16a (optional concat with A16b at splitK; splitK
//    multiple of 32 -> uniform per K-tile).
//  - headMode: B = KH per-head blocks [Kd,64]; head = nbase/64 (uniform).
// Block: 256 threads = 8 waves; tile 256(M) x 64(N); K-step 32.
// Each wave: 32 rows x 64 cols = 8 x v_wmma_f32_16x16x32_bf16 per K-step.
// ---------------------------------------------------------------------------
__global__ __launch_bounds__(256) void gemm_bf16_wmma(
    const float* __restrict__ A32,
    const unsigned short* __restrict__ A16a, int sA,
    const unsigned short* __restrict__ A16b, int sB, int splitK,
    const float* __restrict__ Bw,
    const float* __restrict__ bias,
    unsigned short* __restrict__ Out, int ldo,
    const float* __restrict__ Wred, float redScale, float* __restrict__ redOut,
    int Ncols, int Kd, int do_leaky, int headMode)
{
  __shared__ unsigned short As[2][256][40];   // [buf][row][k] (padded)
  __shared__ unsigned short Bs[2][64][40];    // [buf][col][k] (transposed)

  const int tid   = threadIdx.x;
  const int wave  = tid >> 5;
  const int lane  = tid & 31;
  const int lm    = lane & 15;
  const int lg    = lane >> 4;
  const int rbase = blockIdx.x * 256;
  const int nbase = blockIdx.y * 64;

  // uniform B source selection (per block)
  const float* Bsrc;
  int ldb, cb0;
  if (headMode) { Bsrc = Bw + (size_t)(nbase >> 6) * (size_t)Kd * 64; ldb = 64;    cb0 = 0; }
  else          { Bsrc = Bw;                                          ldb = Ncols; cb0 = nbase; }

  v8f acc[2][4];
  #pragma unroll
  for (int rt = 0; rt < 2; ++rt)
    #pragma unroll
    for (int c = 0; c < 4; ++c)
      #pragma unroll
      for (int r = 0; r < 8; ++r) acc[rt][c][r] = 0.f;

  // staged-in-registers tile (prefetch for double buffering)
  f32x4 ga[8];     // fp32 A row (32 elems)
  u16x8 ga16[4];   // bf16 A row
  f32x4 gb[2];     // 8 B elems

  const int e0  = tid << 3;
  const int kkB = e0 >> 6;          // B k-row for this thread (uniform)
  const int nn0 = e0 & 63;          // first of 8 consecutive B columns

  auto loadRegs = [&](int kt) {
    const int k0 = kt << 5;
    if (A32) {
      const f32x4* src = (const f32x4*)(A32 + (size_t)(rbase + tid) * (size_t)Kd + k0);
      #pragma unroll
      for (int q = 0; q < 8; ++q) ga[q] = src[q];
    } else {
      const unsigned short* Asrc; size_t astr; int aoff;
      if (k0 < splitK) { Asrc = A16a; astr = (size_t)sA; aoff = k0; }
      else             { Asrc = A16b; astr = (size_t)sB; aoff = k0 - splitK; }
      const u16x8* s = (const u16x8*)(Asrc + (size_t)(rbase + tid) * astr + aoff);
      #pragma unroll
      for (int q = 0; q < 4; ++q) ga16[q] = s[q];
    }
    const float* bp = Bsrc + (size_t)(k0 + kkB) * ldb + (cb0 + nn0);
    gb[0] = ((const f32x4*)bp)[0];
    gb[1] = ((const f32x4*)bp)[1];
  };

  auto storeLDS = [&](int p) {
    if (A32) {
      #pragma unroll
      for (int q = 0; q < 4; ++q) {
        const f32x4 fa = ga[2 * q], fb = ga[2 * q + 1];
        u16x8 h;
        h[0] = f2bf(fa[0]); h[1] = f2bf(fa[1]); h[2] = f2bf(fa[2]); h[3] = f2bf(fa[3]);
        h[4] = f2bf(fb[0]); h[5] = f2bf(fb[1]); h[6] = f2bf(fb[2]); h[7] = f2bf(fb[3]);
        ((u16x8*)&As[p][tid][0])[q] = h;
      }
    } else {
      #pragma unroll
      for (int q = 0; q < 4; ++q) ((u16x8*)&As[p][tid][0])[q] = ga16[q];
    }
    #pragma unroll
    for (int q = 0; q < 4; ++q) Bs[p][nn0 + q][kkB]     = f2bf(gb[0][q]);
    #pragma unroll
    for (int q = 0; q < 4; ++q) Bs[p][nn0 + 4 + q][kkB] = f2bf(gb[1][q]);
  };

  const int ksteps = Kd >> 5;
  loadRegs(0);
  int p = 0;
  for (int kt = 0; kt < ksteps; ++kt) {
    storeLDS(p);
    __syncthreads();
    if (kt + 1 < ksteps) loadRegs(kt + 1);   // overlap next global loads

    // A fragments: two 16-row tiles per wave (rows wave*32 + rt*16 + lm)
    v16bf af[2];
    #pragma unroll
    for (int rt = 0; rt < 2; ++rt) {
      const int mr = (wave << 5) + (rt << 4) + lm;
      v16u au;
      #pragma unroll
      for (int v = 0; v < 4; ++v) {
        const int kk = 2 * v + 8 * lg;
        au[2 * v]         = As[p][mr][kk];
        au[2 * v + 1]     = As[p][mr][kk + 1];
        au[8 + 2 * v]     = As[p][mr][16 + kk];
        au[8 + 2 * v + 1] = As[p][mr][16 + kk + 1];
      }
      af[rt] = __builtin_bit_cast(v16bf, au);
    }
    // 4 column tiles, B fragment shared by both row tiles
    #pragma unroll
    for (int c = 0; c < 4; ++c) {
      const int nn = (c << 4) + lm;
      v16u bu;
      #pragma unroll
      for (int e = 0; e < 16; ++e) bu[e] = Bs[p][nn][(lg << 3) + (e < 8 ? e : 8 + e)];
      const v16bf bfv = __builtin_bit_cast(v16bf, bu);
      acc[0][c] = __builtin_amdgcn_wmma_f32_16x16x32_bf16(
          false, af[0], false, bfv, (short)0, acc[0][c], false, false);
      acc[1][c] = __builtin_amdgcn_wmma_f32_16x16x32_bf16(
          false, af[1], false, bfv, (short)0, acc[1][c], false, false);
    }
    p ^= 1;
  }

  if (Wred) {
    // fused [K,1] head: rowsum of leaky(acc+bias)*Wred, shfl-reduce, atomicAdd
    #pragma unroll
    for (int rt = 0; rt < 2; ++rt) {
      float rs[8];
      #pragma unroll
      for (int r = 0; r < 8; ++r) rs[r] = 0.f;
      #pragma unroll
      for (int c = 0; c < 4; ++c) {
        const int col = nbase + (c << 4) + lm;
        const float bv = bias ? bias[col] : 0.f;
        const float wr = Wred[col];
        #pragma unroll
        for (int r = 0; r < 8; ++r) {
          float v = acc[rt][c][r] + bv;
          v = v > 0.f ? v : 0.01f * v;
          rs[r] += v * wr;
        }
      }
      #pragma unroll
      for (int m = 1; m < 16; m <<= 1)
        #pragma unroll
        for (int r = 0; r < 8; ++r)
          rs[r] += __shfl_xor(rs[r], m, 32);
      if (lm == 0) {
        const int rowb = rbase + (wave << 5) + (rt << 4) + (lg << 3);
        #pragma unroll
        for (int r = 0; r < 8; ++r)
          atomicAdd(&redOut[rowb + r], redScale * rs[r]);
      }
    }
  } else {
    // store epilogue: bias + optional leaky, bf16
    #pragma unroll
    for (int rt = 0; rt < 2; ++rt)
      #pragma unroll
      for (int c = 0; c < 4; ++c) {
        const int col = nbase + (c << 4) + lm;
        const float bv = bias ? bias[col] : 0.f;
        #pragma unroll
        for (int r = 0; r < 8; ++r) {
          const int row = rbase + (wave << 5) + (rt << 4) + r + (lg << 3);
          float v = acc[rt][c][r] + bv;
          if (do_leaky) v = v > 0.f ? v : 0.01f * v;
          Out[(size_t)row * ldo + col] = f2bf(v);
        }
      }
  }
}

// ---------------------------------------------------------------------------
// Attention: per batch element, 4 heads, 8x8 self-excluding softmax attention.
// K/Q/V rows are head-concatenated: buf[(n*B+b)*H + k*64 + d].
// Writes other[N,B,H] (bf16) and accumulates logit^2 regularizer.
// ---------------------------------------------------------------------------
__global__ __launch_bounds__(256) void attn_kernel(
    const unsigned short* __restrict__ Qb,
    const unsigned short* __restrict__ Kb,
    const unsigned short* __restrict__ Vb,
    unsigned short* __restrict__ Other,
    float* __restrict__ regs_out)
{
  const int b = blockIdx.x * blockDim.x + threadIdx.x;
  if (b >= BATCH) return;

  float reg_acc[N_AG];
  #pragma unroll
  for (int i = 0; i < N_AG; ++i) reg_acc[i] = 0.f;

  for (int k = 0; k < KH_N; ++k) {
    const int hofs = k * AD_N;

    float lg[N_AG][N_AG];
    #pragma unroll
    for (int i = 0; i < N_AG; ++i) {
      #pragma unroll
      for (int j = 0; j < N_AG; ++j) lg[i][j] = 0.f;
      const u16x8* qrow =
          (const u16x8*)(Qb + ((size_t)i * BATCH + b) * H_DIM + hofs);
      #pragma unroll
      for (int dc = 0; dc < 8; ++dc) {
        const u16x8 q8 = qrow[dc];
        #pragma unroll
        for (int j = 0; j < N_AG; ++j) {
          const u16x8 k8 =
              ((const u16x8*)(Kb + ((size_t)j * BATCH + b) * H_DIM + hofs))[dc];
          float s = 0.f;
          #pragma unroll
          for (int t = 0; t < 8; ++t) s += bf2f(q8[t]) * bf2f(k8[t]);
          lg[i][j] += s;
        }
      }
    }

    // regularizer uses UNSCALED logits, j != i
    #pragma unroll
    for (int i = 0; i < N_AG; ++i)
      #pragma unroll
      for (int j = 0; j < N_AG; ++j)
        if (j != i) reg_acc[i] += lg[i][j] * lg[i][j];

    // softmax (scale 1/sqrt(64), self masked) + probs @ V
    #pragma unroll
    for (int i = 0; i < N_AG; ++i) {
      float pr[N_AG];
      float mx = -1e30f;
      #pragma unroll
      for (int j = 0; j < N_AG; ++j) {
        const float v = (j == i) ? -1e9f : lg[i][j] * 0.125f;
        pr[j] = v;
        mx = fmaxf(mx, v);
      }
      float sum = 0.f;
      #pragma unroll
      for (int j = 0; j < N_AG; ++j) { pr[j] = __expf(pr[j] - mx); sum += pr[j]; }
      const float inv = 1.f / sum;

      float oacc[AD_N];
      #pragma unroll
      for (int d = 0; d < AD_N; ++d) oacc[d] = 0.f;
      #pragma unroll
      for (int j = 0; j < N_AG; ++j) {
        const float pj = pr[j] * inv;
        const u16x8* vrow =
            (const u16x8*)(Vb + ((size_t)j * BATCH + b) * H_DIM + hofs);
        #pragma unroll
        for (int dc = 0; dc < 8; ++dc) {
          const u16x8 v8 = vrow[dc];
          #pragma unroll
          for (int t = 0; t < 8; ++t) oacc[dc * 8 + t] += pj * bf2f(v8[t]);
        }
      }
      unsigned short* orow = Other + ((size_t)i * BATCH + b) * H_DIM + hofs;
      #pragma unroll
      for (int d = 0; d < AD_N; ++d) orow[d] = f2bf(oacc[d]);
    }
  }

  const float cfac = 0.001f / (float)((size_t)BATCH * (N_AG - 1));
  #pragma unroll
  for (int i = 0; i < N_AG; ++i)
    atomicAdd(&regs_out[i], reg_acc[i] * cfac);
}

// out[0..M) = b_c2 - b_b2 (base for fused heads); out[M..M+N_AG) = 0 (regs)
__global__ void init_out(float* __restrict__ out,
                         const float* __restrict__ bc2,
                         const float* __restrict__ bb2)
{
  const int i = blockIdx.x * blockDim.x + threadIdx.x;
  if (i < M_ROWS) out[i] = bc2[0] - bb2[0];
  else if (i < M_ROWS + N_AG) out[i] = 0.f;
}

// ---------------------------------------------------------------------------
extern "C" void kernel_launch(void* const* d_in, const int* in_sizes, int n_in,
                              void* d_out, int out_size, void* d_ws, size_t ws_size,
                              hipStream_t stream)
{
  (void)in_sizes; (void)n_in; (void)out_size; (void)ws_size;
  const float* states = (const float*)d_in[0];
  const float* sa     = (const float*)d_in[1];
  const float* W_enc  = (const float*)d_in[2];
  const float* b_enc  = (const float*)d_in[3];
  const float* W_s    = (const float*)d_in[4];
  const float* b_s    = (const float*)d_in[5];
  const float* W_k    = (const float*)d_in[6];
  const float* W_q    = (const float*)d_in[7];
  const float* W_v    = (const float*)d_in[8];
  const float* b_v    = (const float*)d_in[9];
  const float* W_c1   = (const float*)d_in[10];
  const float* b_c1   = (const float*)d_in[11];
  const float* W_c2   = (const float*)d_in[12];
  const float* b_c2   = (const float*)d_in[13];
  const float* W_b1   = (const float*)d_in[14];
  const float* b_b1   = (const float*)d_in[15];
  const float* W_b2   = (const float*)d_in[16];
  const float* b_b2   = (const float*)d_in[17];
  float* out = (float*)d_out;

  const size_t MH = (size_t)M_ROWS * H_DIM;     // 33.5M bf16 elems per buffer
  unsigned short* ws     = (unsigned short*)d_ws;
  unsigned short* enc_sa = ws;
  unsigned short* enc_s  = ws + 1 * MH;
  unsigned short* Kbuf   = ws + 2 * MH;         // [M, KH*AD] head-concat
  unsigned short* Qbuf   = ws + 3 * MH;
  unsigned short* Vbuf   = ws + 4 * MH;
  unsigned short* Other  = ws + 5 * MH;         // [N,B,H]

  const dim3 blk(256);
  const dim3 grd(M_ROWS / 256, H_DIM / 64);     // (512, 4)

  init_out<<<dim3((M_ROWS + N_AG + 255) / 256), blk, 0, stream>>>(out, b_c2, b_b2);

  // enc_sa = leaky(sa @ W_enc + b_enc)          [131072 x 160] x [160 x 256]
  gemm_bf16_wmma<<<grd, blk, 0, stream>>>(
      sa, nullptr, 0, nullptr, 0, S_DIM + A_DIM,
      W_enc, b_enc, enc_sa, H_DIM,
      nullptr, 0.f, nullptr, H_DIM, S_DIM + A_DIM, 1, 0);

  // enc_s = leaky(states @ W_s + b_s)           [131072 x 128] x [128 x 256]
  gemm_bf16_wmma<<<grd, blk, 0, stream>>>(
      states, nullptr, 0, nullptr, 0, S_DIM,
      W_s, b_s, enc_s, H_DIM,
      nullptr, 0.f, nullptr, H_DIM, S_DIM, 1, 0);

  // fused per-head projections (headMode): one pass over enc per K/Q/V
  gemm_bf16_wmma<<<grd, blk, 0, stream>>>(          // K = enc_sa @ W_k[head]
      nullptr, enc_sa, H_DIM, nullptr, 0, H_DIM,
      W_k, nullptr, Kbuf, H_DIM,
      nullptr, 0.f, nullptr, H_DIM, H_DIM, 0, 1);
  gemm_bf16_wmma<<<grd, blk, 0, stream>>>(          // Q = enc_s @ W_q[head]
      nullptr, enc_s, H_DIM, nullptr, 0, H_DIM,
      W_q, nullptr, Qbuf, H_DIM,
      nullptr, 0.f, nullptr, H_DIM, H_DIM, 0, 1);
  gemm_bf16_wmma<<<grd, blk, 0, stream>>>(          // V = leaky(enc_sa @ W_v + b_v)
      nullptr, enc_sa, H_DIM, nullptr, 0, H_DIM,
      W_v, b_v, Vbuf, H_DIM,
      nullptr, 0.f, nullptr, H_DIM, H_DIM, 1, 1);

  // softmax attention + regularizer
  attn_kernel<<<dim3(BATCH / 256), blk, 0, stream>>>(
      Qbuf, Kbuf, Vbuf, Other, out + M_ROWS);

  // critic: out += leaky([enc_sa|other] @ W_c1 + b_c1) @ W_c2   (fused head)
  gemm_bf16_wmma<<<grd, blk, 0, stream>>>(
      nullptr, enc_sa, H_DIM, Other, H_DIM, H_DIM,
      W_c1, b_c1, nullptr, 0,
      W_c2, 1.0f, out, H_DIM, 2 * H_DIM, 1, 0);

  // baseline: out -= leaky(enc_s @ W_b1 + b_b1) @ W_b2          (fused head)
  gemm_bf16_wmma<<<grd, blk, 0, stream>>>(
      nullptr, enc_s, H_DIM, nullptr, 0, H_DIM,
      W_b1, b_b1, nullptr, 0,
      W_b2, -1.0f, out, H_DIM, H_DIM, 1, 0);
}
